// MRALSTM_46969762349094
// MI455X (gfx1250) — matrix operand order repaired and checked by reading, not compiled, not run on hardware
//
#include <hip/hip_runtime.h>
#include <hip/hip_bf16.h>
#include <stdint.h>
#include <stddef.h>

// ---------------------------------------------------------------------------
// MRA-LSTM forward for MI455X (gfx1250, wave32, WMMA).
// GEMMs run as bf16 V_WMMA_F32_16X16X32_BF16 with fp32 accumulate; weights are
// packed to bf16 once per call (L2-resident: ~12MB), activations keep an fp32
// master + bf16 mirror. Branch selection is wave-uniform and resolved on
// device so the launch graph is static (graph-capture safe).
// This revision forces addrspace(1) (global) on all memory traffic so the
// backend emits global_load/global_store instead of flat_* (flat ties up the
// LDS datapath and double-counts against LOADcnt+DScnt on CDNA5).
// ---------------------------------------------------------------------------

typedef __attribute__((ext_vector_type(16))) __bf16 v16bf;
typedef __attribute__((ext_vector_type(8)))  float  v8f;
typedef __attribute__((ext_vector_type(4)))  unsigned int u32x4;

#define AS1 __attribute__((address_space(1)))

#define HH 512      // hidden
#define BB 512      // batch
#define TT 128      // seq len
#define NG 2048     // 4 gates * H
#define OUTD 64

__device__ __forceinline__ float sigf(float x) { return 1.0f / (1.0f + __expf(-x)); }

// ---------------- weight family packing: 4x(H x H) fp32 -> (2048 x H) bf16 ---
__global__ void pack_fam_kernel(const float* wf, const float* wi, const float* wo, const float* wg,
                                const float* bfv, const float* biv, const float* bov, const float* bgv,
                                unsigned short* dstW, float* dstB)
{
    size_t i = (size_t)blockIdx.x * blockDim.x + threadIdx.x;
    if (i < (size_t)NG * HH) {
        int row = (int)(i / HH), k = (int)(i % HH);
        int gate = row >> 9, rr = row & 511;
        const float* s = gate == 0 ? wf : gate == 1 ? wi : gate == 2 ? wo : wg;
        __bf16 v = (__bf16)((AS1 const float*)s)[(size_t)rr * HH + k];
        ((AS1 unsigned short*)dstW)[i] = __builtin_bit_cast(unsigned short, v);
    }
    if (i < NG) {
        int gate = (int)i >> 9, rr = (int)i & 511;
        const float* s = gate == 0 ? bfv : gate == 1 ? biv : gate == 2 ? bov : bgv;
        ((AS1 float*)dstB)[i] = ((AS1 const float*)s)[rr];
    }
}

// ---------------- branch select: idx = 2*(mean(sp)>=.5) + (mean(sl)>=.5) ----
__global__ void select_kernel(const float* sp, const float* sl, int* idx_out)
{
    __shared__ float r0[256], r1[256];
    int tid = threadIdx.x;
    AS1 const float* gsp = (AS1 const float*)sp;
    r0[tid] = gsp[tid] + gsp[tid + 256];
    if (sl) {
        AS1 const float* gsl = (AS1 const float*)sl;
        r1[tid] = gsl[tid] + gsl[tid + 256];
    } else {
        r1[tid] = 0.0f;
    }
    __syncthreads();
    for (int o = 128; o; o >>= 1) {
        if (tid < o) { r0[tid] += r0[tid + o]; r1[tid] += r1[tid + o]; }
        __syncthreads();
    }
    if (tid == 0) {
        float mp = r0[0] * (1.0f / 512.0f), ml = r1[0] * (1.0f / 512.0f);
        *(AS1 int*)idx_out = ((mp >= 0.5f) ? 2 : 0) + ((ml >= 0.5f) ? 1 : 0);
    }
}

// ---------------- fused gate GEMM:  G = sum_terms act(A_t @ W_t^T + b_t) -----
// act = sigmoid for f/i/o columns (n<1536), tanh for g columns (n>=1536).
struct GemmArgs {
    const int* idx_p; unsigned mask;                 // 4-bit active-term mask per idx
    const unsigned short *A0, *A1, *A2;              // bf16 activations (B x 512)
    const unsigned short *W0, *W1, *W2;              // bf16 fams (2048 x 512)
    const float *B0, *B1, *B2;                       // fp32 fam biases (2048)
    float* G;                                        // fp32 (B x 2048)
};

__global__ __launch_bounds__(256) void gemm_gates_kernel(GemmArgs a)
{
    int idx = *(AS1 const int*)a.idx_p;              // wave-uniform
    unsigned active = (a.mask >> (idx * 4)) & 0xFu;
    if (!active) return;                             // uniform early-exit

    const unsigned short* Aarr[3] = { a.A0, a.A1, a.A2 };
    const unsigned short* Warr[3] = { a.W0, a.W1, a.W2 };
    const float*          Barr[3] = { a.B0, a.B1, a.B2 };

    int lane  = threadIdx.x & 31;
    int wave  = threadIdx.x >> 5;
    int m0    = blockIdx.x * 64  + (wave & 3) * 16;  // batch rows
    int n0    = blockIdx.y * 128 + (wave >> 2) * 64; // gate cols
    int lrow  = lane & 15;
    int khalf = lane >> 4;

    float sum[4][8];
#pragma unroll
    for (int tn = 0; tn < 4; ++tn)
#pragma unroll
        for (int r = 0; r < 8; ++r) sum[tn][r] = 0.0f;

    for (int t = 0; t < 3; ++t) {
        if (!((active >> t) & 1)) continue;          // uniform
        // global (addrspace 1) bases, indexed in u32x4 (8 x bf16) units ->
        // 32-bit offsets + saddr-form global_load_b128.
        AS1 const u32x4* Ab   = (AS1 const u32x4*)Aarr[t];
        AS1 const u32x4* Wb   = (AS1 const u32x4*)Warr[t];
        AS1 const float* bias = (AS1 const float*)Barr[t];
        int aRow = ((m0 + lrow) * HH + khalf * 8) >> 3;     // + kk/8 each iter
        v8f acc[4] = {};
        for (int kk = 0; kk < HH; kk += 32) {
            // A fragment (16x32 bf16): lane holds row m0+lrow, K chunks at
            // khalf*8 .. +7 and khalf*8+16 .. +23 (ISA 16-bit A layout).
            union { v16bf v; u32x4 q[2]; } ua;
            int ai = aRow + (kk >> 3);
            ua.q[0] = Ab[ai];
            ua.q[1] = Ab[ai + 2];
#pragma unroll
            for (int tn = 0; tn < 4; ++tn) {
                // B fragment (32x16 bf16): lane holds output col n, K chunks at
                // kk + khalf*8 .. +7 and kk + khalf*8+16 .. +23 (matches A).
                int ocol = n0 + tn * 16 + lrow;
                int bi = (ocol * HH + kk + khalf * 8) >> 3;
                union { v16bf v; u32x4 q[2]; } ub;
                ub.q[0] = Wb[bi];
                ub.q[1] = Wb[bi + 2];
                acc[tn] = __builtin_amdgcn_wmma_f32_16x16x32_bf16(
                    false, ua.v, false, ub.v, (short)0, acc[tn], false, false);
            }
        }
        // per-term epilogue: bias + inner activation, then accumulate
#pragma unroll
        for (int tn = 0; tn < 4; ++tn) {
            int n = n0 + tn * 16 + lrow;
            float bv = bias[n];
            bool isT = (n >= 1536);
#pragma unroll
            for (int r = 0; r < 8; ++r) {
                float v = acc[tn][r] + bv;
                sum[tn][r] += isT ? tanhf(v) : sigf(v);
            }
        }
    }
    // C/D layout: vgpr r -> M = m0 + r + 8*khalf, N = n0 + tn*16 + lrow
    AS1 float* Gg = (AS1 float*)a.G;
#pragma unroll
    for (int tn = 0; tn < 4; ++tn) {
        int n = n0 + tn * 16 + lrow;
#pragma unroll
        for (int r = 0; r < 8; ++r) {
            int m = m0 + r + 8 * khalf;
            Gg[(size_t)m * NG + n] = sum[tn][r];
        }
    }
}

// ---------------- pointwise state update + boundary logit ------------------
// gateMode nibble: 0=zero, 1=outer(sig f/i/o, g raw), 2=raw(f=0), 3=f=0 +
// outer sig on i/o, 4=const(i=o=0.5, f=g=0).
struct UpdateArgs {
    const int* idx_p;
    unsigned gateMode;                // nibble per idx
    unsigned slogMask;                // 4-bit active slog-term mask per idx
    const float* G;
    float* h; float* c; unsigned short* hb; float* s;
    const float *SA0, *SA1, *SA2;     // fp32 h inputs for slog dots (B x 512)
    const float *SW0, *SW1, *SW2;     // s-weight vectors (512)
    const float *SB0, *SB1, *SB2;     // s-bias scalars
};

__global__ __launch_bounds__(256) void update_kernel(UpdateArgs a)
{
    int b = blockIdx.x, tid = threadIdx.x;
    int idx = *(AS1 const int*)a.idx_p;
    unsigned sm = (a.slogMask >> (idx * 4)) & 0xFu;
    int gm = (int)((a.gateMode >> (idx * 4)) & 0xFu);

    const float* SA[3] = { a.SA0, a.SA1, a.SA2 };
    const float* SW[3] = { a.SW0, a.SW1, a.SW2 };
    const float* SB[3] = { a.SB0, a.SB1, a.SB2 };

    __shared__ float red[256];
    float part = 0.0f;
    for (int t = 0; t < 3; ++t) {
        if (!((sm >> t) & 1)) continue;
        AS1 const float* A = (AS1 const float*)SA[t];
        AS1 const float* W = (AS1 const float*)SW[t];
        part += A[(size_t)b * HH + tid] * W[tid]
              + A[(size_t)b * HH + tid + 256] * W[tid + 256];
        if (tid == 0) part += *(AS1 const float*)SB[t];
    }
    red[tid] = part;
    __syncthreads();
    for (int o = 128; o; o >>= 1) {
        if (tid < o) red[tid] += red[tid + o];
        __syncthreads();
    }
    float slog = red[0];

    bool sp = (idx >> 1) & 1, sl = idx & 1;
    AS1 const float* Gr = (AS1 const float*)(a.G) + (size_t)b * NG;
    AS1 float* hP = (AS1 float*)a.h;
    AS1 float* cP = (AS1 float*)a.c;
    AS1 unsigned short* hbP = (AS1 unsigned short*)a.hb;
    for (int j = tid; j < HH; j += 256) {
        float Gf = 0, Gi = 0, Go = 0, Gg = 0;
        if (gm == 1 || gm == 2 || gm == 3) {
            Gf = Gr[j]; Gi = Gr[512 + j]; Go = Gr[1024 + j]; Gg = Gr[1536 + j];
        }
        float f, i, o, g;
        switch (gm) {
            case 1:  f = sigf(Gf); i = sigf(Gi); o = sigf(Go); g = Gg; break;
            case 2:  f = 0.0f;     i = Gi;       o = Go;       g = Gg; break;
            case 3:  f = 0.0f;     i = sigf(Gi); o = sigf(Go); g = Gg; break;
            case 4:  f = 0.0f;     i = 0.5f;     o = 0.5f;     g = 0.0f; break;
            default: f = 0.0f;     i = 0.0f;     o = 0.0f;     g = 0.0f; break;
        }
        float cp = cP[(size_t)b * HH + j];
        float cn = sp ? (i * g) : (f * cp + i * g);
        float hp = hP[(size_t)b * HH + j];
        float hn = (!sp || !sl) ? o * tanhf(cn) : hp;
        cP[(size_t)b * HH + j] = cn;
        hP[(size_t)b * HH + j] = hn;
        hbP[(size_t)b * HH + j] = __builtin_bit_cast(unsigned short, (__bf16)hn);
    }
    if (tid == 0) ((AS1 float*)a.s)[b] = (slog > 0.5f) ? 1.0f : 0.0f;  // STE fwd
}

// ---------------- hierarchical state excitation + output head --------------
__global__ __launch_bounds__(256) void hse_out_kernel(
    const float* h0, const float* h1, const float* h2,
    const float* q0w, const float* q0b, const float* q1w, const float* q1b,
    const float* q2w, const float* q2b, const float* ow, const float* ob, float* out)
{
    int b = blockIdx.x, tid = threadIdx.x;
    __shared__ float red[256];
    __shared__ float rk[3];
    __shared__ float e[HH];
    AS1 const float* hs[3] = { (AS1 const float*)h0, (AS1 const float*)h1,
                               (AS1 const float*)h2 };
    AS1 const float* qw[3] = { (AS1 const float*)q0w, (AS1 const float*)q1w,
                               (AS1 const float*)q2w };
    AS1 const float* qb[3] = { (AS1 const float*)q0b, (AS1 const float*)q1b,
                               (AS1 const float*)q2b };
    for (int k = 0; k < 3; ++k) {
        float part = 0.0f;
        for (int jj = tid; jj < 3 * HH; jj += 256) {
            int seg = jj >> 9, j = jj & 511;
            part += qw[k][jj] * hs[seg][(size_t)b * HH + j];
        }
        red[tid] = part;
        __syncthreads();
        for (int o = 128; o; o >>= 1) {
            if (tid < o) red[tid] += red[tid + o];
            __syncthreads();
        }
        if (tid == 0) rk[k] = sigf(red[0] + qb[k][0]);
        __syncthreads();
    }
    for (int j = tid; j < HH; j += 256) {
        float v = rk[0] * hs[0][(size_t)b * HH + j]
                + rk[1] * hs[1][(size_t)b * HH + j]
                + rk[2] * hs[2][(size_t)b * HH + j];
        e[j] = v > 0.0f ? v : 0.0f;
    }
    __syncthreads();
    if (tid < OUTD) {
        AS1 const float* owg = (AS1 const float*)ow;
        float acc = ((AS1 const float*)ob)[tid];
        for (int j = 0; j < HH; ++j) acc += owg[(size_t)tid * HH + j] * e[j];
        ((AS1 float*)out)[(size_t)b * OUTD + tid] = acc;
    }
}

// ---------------------------------------------------------------------------
extern "C" void kernel_launch(void* const* d_in, const int* in_sizes, int n_in,
                              void* d_out, int out_size, void* d_ws, size_t ws_size,
                              hipStream_t stream)
{
    (void)in_sizes; (void)n_in; (void)out_size;
    // Input flattening assumed in setup_inputs() insertion order:
    // [0]=x; cell0: W_f..W_s, U_f..U_s, Z_f..Z_s (w,b each) -> 1..30;
    // cell1: W,V,Z -> 31..60; cell2: W,V -> 61..80; Qr -> 81..86; out -> 87..88.
    auto F = [&](int i) { return (const float*)d_in[i]; };

    const size_t FAMW = (size_t)NG * HH * 2;   // bf16 fam
    const size_t FAMB = (size_t)NG * 4;        // fp32 fam bias
    const size_t GB   = (size_t)BB * NG * 4;   // gate buffer
    const size_t HB   = (size_t)BB * HH * 4;   // fp32 state
    const size_t HBB  = (size_t)BB * HH * 2;   // bf16 mirror
    const size_t SB_  = (size_t)BB * 4;        // boundary state

    char* ws = (char*)d_ws;
    size_t off = 0;
    auto alloc = [&](size_t b) { void* p = ws + off; off += (b + 255) & ~(size_t)255; return p; };

    unsigned short* fam[6]; float* famb[6];
    for (int f = 0; f < 6; ++f) fam[f]  = (unsigned short*)alloc(FAMW);
    for (int f = 0; f < 6; ++f) famb[f] = (float*)alloc(FAMB);
    float* G = (float*)alloc(GB);
    char* zeroStart = ws + off;
    float* h[3]; float* c[3]; unsigned short* hb[3]; float* s[3];
    for (int k = 0; k < 3; ++k) h[k]  = (float*)alloc(HB);
    for (int k = 0; k < 3; ++k) c[k]  = (float*)alloc(HB);
    for (int k = 0; k < 3; ++k) hb[k] = (unsigned short*)alloc(HBB);
    for (int k = 0; k < 3; ++k) s[k]  = (float*)alloc(SB_);
    size_t zeroBytes = (size_t)((ws + off) - zeroStart);
    int* idxp[3];
    for (int k = 0; k < 3; ++k) idxp[k] = (int*)alloc(16);
    if (off > ws_size) return;  // not enough scratch; bail deterministically

    // --- pack the 6 reachable weight families to bf16 ---
    // fam order: 0=Z(cell0), 1=W(cell1), 2=V(cell1), 3=Z(cell1), 4=W(cell2), 5=V(cell2)
    const int wbase[6] = { 21, 31, 41, 51, 61, 71 };
    dim3 pgrid((unsigned)(((size_t)NG * HH + 255) / 256));
    for (int f = 0; f < 6; ++f) {
        int wb = wbase[f];
        hipLaunchKernelGGL(pack_fam_kernel, pgrid, dim3(256), 0, stream,
            F(wb), F(wb + 2), F(wb + 4), F(wb + 6),
            F(wb + 1), F(wb + 3), F(wb + 5), F(wb + 7),
            fam[f], famb[f]);
    }
    hipMemsetAsync(zeroStart, 0, zeroBytes, stream);

    // --- per-layer static configs (branch resolved on device) ---
    GemmArgs ga[3]; UpdateArgs ua[3];
    // k=2: br_A -> act(W2 h2)+act(V2 h1)
    ga[2] = { nullptr, 0x0030u, hb[2], hb[1], nullptr, fam[4], fam[5], nullptr,
              famb[4], famb[5], nullptr, G };
    // k=1: br_A -> W1 h1 + V1 h0 ; br_B/C -> Z1 h2
    ga[1] = { nullptr, 0x4430u, hb[1], hb[0], hb[2], fam[1], fam[2], fam[3],
              famb[1], famb[2], famb[3], G };
    // k=0: br_B -> Z0 h1 (only reachable gate branch)
    ga[0] = { nullptr, 0x0100u, hb[1], nullptr, nullptr, fam[0], nullptr, nullptr,
              famb[0], nullptr, nullptr, G };
    for (int k = 0; k < 3; ++k) ga[k].idx_p = idxp[k];

    // slog terms: (A, w, b) triples; masks per idx.
    ua[2] = { idxp[2], 0x4010u, 0x0232u, G, h[2], c[2], hb[2], s[2],
              h[2], h[1], nullptr, F(69), F(79), nullptr, F(70), F(80), nullptr };
    ua[1] = { idxp[1], 0x3210u, 0x4432u, G, h[1], c[1], hb[1], s[1],
              h[1], h[0], h[2], F(39), F(49), F(59), F(40), F(50), F(60) };
    ua[0] = { idxp[0], 0x0200u, 0x0201u, G, h[0], c[0], hb[0], s[0],
              h[0], h[1], nullptr, F(9), F(29), nullptr, F(10), F(30), nullptr };

    // --- sequential scan: top-down layer update each timestep ---
    dim3 ggrid(BB / 64, NG / 128);
    for (int t = 0; t < TT; ++t) {
        for (int k = 2; k >= 0; --k) {
            hipLaunchKernelGGL(select_kernel, dim3(1), dim3(256), 0, stream,
                               s[k], (k > 0) ? s[k - 1] : nullptr, idxp[k]);
            hipLaunchKernelGGL(gemm_gates_kernel, ggrid, dim3(256), 0, stream, ga[k]);
            hipLaunchKernelGGL(update_kernel, dim3(BB), dim3(256), 0, stream, ua[k]);
        }
    }

    // --- HSE + output projection ---
    hipLaunchKernelGGL(hse_out_kernel, dim3(BB), dim3(256), 0, stream,
        h[0], h[1], h[2],
        F(81), F(82), F(83), F(84), F(85), F(86),
        F(87), F(88), (float*)d_out);
}
